// HierarchicalRiskPropagationNetwork_9715216023605
// MI455X (gfx1250) — matrix operand order repaired and checked by reading, not compiled, not run on hardware
//
#include <hip/hip_runtime.h>
#include <math.h>

typedef _Float16 v16h __attribute__((ext_vector_type(16)));
typedef float    v8f  __attribute__((ext_vector_type(8)));

#define NND 3072   // nodes / sequence length
#define NB  8      // graphs
#define NC  6      // risk categories

__device__ __forceinline__ float sigmf(float x){ return 1.0f/(1.0f+expf(-x)); }

// ============================================================================
// Generic f16-WMMA GEMM:  C[M,N] = act( A[M,K] @ W[N,K]^T + bias (+bias2) )
// One wave computes a 16x64 strip of C (4 WMMA accumulators); the A operand is
// loaded once per k-step and reused by 4 v_wmma issues. N % 64 == 0, K % 32 == 0.
// ============================================================================
__global__ __launch_bounds__(32) void k_wmma_gemm(
    const float* __restrict__ A, const float* __restrict__ W,
    const float* __restrict__ bias, const float* __restrict__ bias2,
    float* __restrict__ C, int M, int N, int K, int act)
{
  (void)M;
  const int lane = threadIdx.x;
  const int mn   = lane & 15;
  const int lh   = lane >> 4;
  const int mbase = blockIdx.x * 16;
  const int nbase = blockIdx.y * 64;

  v8f acc[4];
  #pragma unroll
  for (int nt = 0; nt < 4; ++nt)
    #pragma unroll
    for (int r = 0; r < 8; ++r) acc[nt][r] = 0.0f;

  const float* ap  = A + (size_t)(mbase + mn) * K;
  const float* wp0 = W + (size_t)(nbase +  0 + mn) * K;
  const float* wp1 = W + (size_t)(nbase + 16 + mn) * K;
  const float* wp2 = W + (size_t)(nbase + 32 + mn) * K;
  const float* wp3 = W + (size_t)(nbase + 48 + mn) * K;

  for (int kb = 0; kb < K; kb += 32) {
    v16h a;
    #pragma unroll
    for (int v = 0; v < 8; ++v) {
      const int ka = ((v >> 2) << 4) + lh * 8 + ((v & 3) << 1);   // A: 16x32 layout
      a[2*v]   = (_Float16)ap[kb + ka];
      a[2*v+1] = (_Float16)ap[kb + ka + 1];
    }
    v16h b0, b1, b2, b3;
    #pragma unroll
    for (int v = 0; v < 8; ++v) {
      const int kw = ((v >> 2) << 4) + lh * 8 + ((v & 3) << 1);   // B: 32x16 layout
      b0[2*v] = (_Float16)wp0[kb + kw]; b0[2*v+1] = (_Float16)wp0[kb + kw + 1];
      b1[2*v] = (_Float16)wp1[kb + kw]; b1[2*v+1] = (_Float16)wp1[kb + kw + 1];
      b2[2*v] = (_Float16)wp2[kb + kw]; b2[2*v+1] = (_Float16)wp2[kb + kw + 1];
      b3[2*v] = (_Float16)wp3[kb + kw]; b3[2*v+1] = (_Float16)wp3[kb + kw + 1];
    }
    acc[0] = __builtin_amdgcn_wmma_f32_16x16x32_f16(false, a, false, b0, (short)0, acc[0], false, false);
    acc[1] = __builtin_amdgcn_wmma_f32_16x16x32_f16(false, a, false, b1, (short)0, acc[1], false, false);
    acc[2] = __builtin_amdgcn_wmma_f32_16x16x32_f16(false, a, false, b2, (short)0, acc[2], false, false);
    acc[3] = __builtin_amdgcn_wmma_f32_16x16x32_f16(false, a, false, b3, (short)0, acc[3], false, false);
  }

  #pragma unroll
  for (int nt = 0; nt < 4; ++nt) {
    const int col = nbase + nt * 16 + mn;
    float bv = 0.0f;
    if (bias)  bv += bias[col];
    if (bias2) bv += bias2[col];
    #pragma unroll
    for (int r = 0; r < 8; ++r) {
      float x = acc[nt][r] + bv;
      if (act) x = fmaxf(x, 0.0f);
      C[(size_t)(mbase + r + 8 * lh) * N + col] = x;
    }
  }
}

// ============================================================================
// Flash attention. qkv: [3072][768] f32 (q|k|v, 4 heads x 64). oat: [3072][256].
// 4 waves per block share one LDS K/V stage (64 queries / block); each wave
// owns a 16-query tile and a private P-transpose slice in LDS.
// ============================================================================
__device__ __forceinline__ v16h ldsB(const _Float16* p, int lh)
{ // B operand gather: K pairs in table order (matches A)
  v16h b;
  #pragma unroll
  for (int v = 0; v < 8; ++v) {
    const int kk = ((v >> 2) << 4) + lh * 8 + ((v & 3) << 1);
    b[2*v]   = p[kk];
    b[2*v+1] = p[kk + 1];
  }
  return b;
}

__global__ __launch_bounds__(128) void k_flash_attn(
    const float* __restrict__ qkv, float* __restrict__ oat)
{
  __shared__ _Float16 Ksh[32 * 64];        // [key][d]
  __shared__ _Float16 Vsh[64 * 32];        // [d][key] (transposed)
  __shared__ _Float16 Psh[4 * 16 * 32];    // per-wave [q][key]

  const int tid  = threadIdx.x;
  const int wave = tid >> 5;
  const int lane = tid & 31;
  const int mn   = lane & 15;
  const int lh   = lane >> 4;
  const int head  = blockIdx.y;
  const int qbase = blockIdx.x * 64 + wave * 16;
  _Float16* Pw = Psh + wave * 512;

  // Q in A-operand layout, scale 1/sqrt(64) folded in (exact in f16)
  v16h aq0, aq1;
  {
    const float* qp = qkv + (size_t)(qbase + mn) * 768 + head * 64;
    #pragma unroll
    for (int v = 0; v < 8; ++v) {
      const int k = ((v >> 2) << 4) + lh * 8 + ((v & 3) << 1);
      aq0[2*v]   = (_Float16)(qp[k]          * 0.125f);
      aq0[2*v+1] = (_Float16)(qp[k + 1]      * 0.125f);
      aq1[2*v]   = (_Float16)(qp[32 + k]     * 0.125f);
      aq1[2*v+1] = (_Float16)(qp[32 + k + 1] * 0.125f);
    }
  }

  v8f o0, o1, o2, o3;
  float mrow[8], lrow[8];
  #pragma unroll
  for (int r = 0; r < 8; ++r) { o0[r]=o1[r]=o2[r]=o3[r]=0.0f; mrow[r]=-1e30f; lrow[r]=0.0f; }

  const int srow = tid >> 2;        // 0..31  (key row staged by this thread)
  const int scol = (tid & 3) * 16;  // 0,16,32,48 (d segment)

  for (int kc = 0; kc < NND; kc += 32) {
    { // cooperative stage of K (row major) and V (transposed)
      const float* kp = qkv + (size_t)(kc + srow) * 768 + 256 + head * 64 + scol;
      const float* vp = qkv + (size_t)(kc + srow) * 768 + 512 + head * 64 + scol;
      #pragma unroll
      for (int j = 0; j < 16; ++j) {
        Ksh[srow * 64 + scol + j] = (_Float16)kp[j];
        Vsh[(scol + j) * 32 + srow] = (_Float16)vp[j];
      }
      if (kc + 32 < NND) {              // prefetch next chunk (global_prefetch_b8)
        __builtin_prefetch(kp + 32 * 768, 0, 3);
        __builtin_prefetch(vp + 32 * 768, 0, 3);
      }
    }
    __syncthreads();

    // S = Q K^T (two 16x16 key tiles)
    v8f s0, s1;
    #pragma unroll
    for (int r = 0; r < 8; ++r) { s0[r] = 0.0f; s1[r] = 0.0f; }
    s0 = __builtin_amdgcn_wmma_f32_16x16x32_f16(false, aq0, false, ldsB(Ksh + mn * 64,             lh), (short)0, s0, false, false);
    s0 = __builtin_amdgcn_wmma_f32_16x16x32_f16(false, aq1, false, ldsB(Ksh + mn * 64 + 32,        lh), (short)0, s0, false, false);
    s1 = __builtin_amdgcn_wmma_f32_16x16x32_f16(false, aq0, false, ldsB(Ksh + (16 + mn) * 64,      lh), (short)0, s1, false, false);
    s1 = __builtin_amdgcn_wmma_f32_16x16x32_f16(false, aq1, false, ldsB(Ksh + (16 + mn) * 64 + 32, lh), (short)0, s1, false, false);

    // online softmax; row r of this thread = qbase + r + 8*lh
    #pragma unroll
    for (int r = 0; r < 8; ++r) {
      float mx = fmaxf(s0[r], s1[r]);
      #pragma unroll
      for (int off = 8; off >= 1; off >>= 1) mx = fmaxf(mx, __shfl_xor(mx, off, 32));
      const float mnew = fmaxf(mrow[r], mx);
      const float sc = expf(mrow[r] - mnew);
      const float p0 = expf(s0[r] - mnew);
      const float p1 = expf(s1[r] - mnew);
      float ps = p0 + p1;
      #pragma unroll
      for (int off = 8; off >= 1; off >>= 1) ps += __shfl_xor(ps, off, 32);
      lrow[r] = lrow[r] * sc + ps;
      mrow[r] = mnew;
      o0[r] *= sc; o1[r] *= sc; o2[r] *= sc; o3[r] *= sc;
      const int m = r + 8 * lh;                 // C-layout row -> P row
      Pw[m * 32 + mn]      = (_Float16)p0;
      Pw[m * 32 + 16 + mn] = (_Float16)p1;
    }
    __syncthreads();

    // O += P V   (P re-gathered in A layout from this wave's LDS slice)
    v16h pa;
    #pragma unroll
    for (int v = 0; v < 8; ++v) {
      const int k = ((v >> 2) << 4) + lh * 8 + ((v & 3) << 1);
      pa[2*v]   = Pw[mn * 32 + k];
      pa[2*v+1] = Pw[mn * 32 + k + 1];
    }
    o0 = __builtin_amdgcn_wmma_f32_16x16x32_f16(false, pa, false, ldsB(Vsh + mn * 32,        lh), (short)0, o0, false, false);
    o1 = __builtin_amdgcn_wmma_f32_16x16x32_f16(false, pa, false, ldsB(Vsh + (16 + mn) * 32, lh), (short)0, o1, false, false);
    o2 = __builtin_amdgcn_wmma_f32_16x16x32_f16(false, pa, false, ldsB(Vsh + (32 + mn) * 32, lh), (short)0, o2, false, false);
    o3 = __builtin_amdgcn_wmma_f32_16x16x32_f16(false, pa, false, ldsB(Vsh + (48 + mn) * 32, lh), (short)0, o3, false, false);
    __syncthreads();
  }

  #pragma unroll
  for (int r = 0; r < 8; ++r) {
    const int row = qbase + r + 8 * lh;
    const float inv = 1.0f / lrow[r];
    float* op = oat + (size_t)row * 256 + head * 64 + mn;
    op[0]  = o0[r] * inv;
    op[16] = o1[r] * inv;
    op[32] = o2[r] * inv;
    op[48] = o3[r] * inv;
  }
}

// ============================================================================
// BiLSTM scan: grid=2 (dir), block=1024. pre_*: [3072][1024], hout: [3072][512]
// ============================================================================
__global__ __launch_bounds__(1024) void k_lstm(
    const float* __restrict__ pre_f, const float* __restrict__ pre_r,
    const float* __restrict__ whhT, float* __restrict__ hout)
{
  __shared__ float hsh[256];
  __shared__ float gsh[1024];
  const int j = threadIdx.x;
  const int dir = blockIdx.x;
  const float* pre = dir ? pre_r : pre_f;
  const float* wT  = whhT + (size_t)dir * 262144;   // [256][1024]
  if (j < 256) hsh[j] = 0.0f;
  float c = 0.0f;
  __syncthreads();
  for (int s = 0; s < NND; ++s) {
    const int t = dir ? (NND - 1 - s) : s;
    float g = pre[(size_t)t * 1024 + j];
    for (int k = 0; k < 256; ++k) g += wT[(size_t)k * 1024 + j] * hsh[k];
    gsh[j] = g;
    __syncthreads();
    if (j < 256) {
      const float ig = sigmf(gsh[j]);
      const float fg = sigmf(gsh[256 + j]);
      const float gg = tanhf(gsh[512 + j]);
      const float og = sigmf(gsh[768 + j]);
      c = fg * c + ig * gg;
      const float h = og * tanhf(c);
      hsh[j] = h;
      hout[(size_t)t * 512 + dir * 256 + j] = h;
    }
    __syncthreads();
  }
}

// ============================================================================
// small helper kernels
// ============================================================================
__global__ void k_bias_sum(const float* a, const float* b, const float* c2, const float* d2, float* out) {
  const int i = blockIdx.x * blockDim.x + threadIdx.x;
  if (i < 1024)      out[i] = a[i] + b[i];
  else if (i < 2048) out[i] = c2[i - 1024] + d2[i - 1024];
}

__global__ void k_transpose_whh(const float* whf, const float* whr, float* whhT) {
  const int idx = blockIdx.x * blockDim.x + threadIdx.x;   // < 524288
  const int dir = idx >> 18;
  const int e   = idx & 262143;
  const int k   = e >> 10;
  const int j   = e & 1023;
  const float* w = dir ? whr : whf;
  whhT[idx] = w[j * 256 + k];
}

__global__ void k_deg_init(float* deg) {
  const int i = blockIdx.x * blockDim.x + threadIdx.x;
  if (i < NND) deg[i] = 1.0f;                              // self loop
}
__global__ void k_deg_count(const int* __restrict__ ei, int E, float* deg) {
  const int e = blockIdx.x * blockDim.x + threadIdx.x;
  if (e < E) atomicAdd(&deg[ei[E + e]], 1.0f);
}
__global__ void k_deg_rsqrt(float* deg) {
  const int i = blockIdx.x * blockDim.x + threadIdx.x;
  if (i < NND) deg[i] = rsqrtf(deg[i]);
}
__global__ void k_agg_self(const float* __restrict__ xw, const float* __restrict__ dinv, float* agg) {
  const int t = blockIdx.x, c = threadIdx.x;
  agg[t * 256 + c] = xw[t * 256 + c] * dinv[t] * dinv[t];
}
__global__ void k_agg_edges(const int* __restrict__ ei, int E, const float* __restrict__ dinv,
                            const float* __restrict__ xw, float* agg) {
  const int e = blockIdx.x, c = threadIdx.x;
  const int s = ei[e], d = ei[E + e];
  atomicAdd(&agg[d * 256 + c], xw[s * 256 + c] * (dinv[s] * dinv[d]));
}
__global__ void k_gcn_fin(const float* __restrict__ agg, const float* __restrict__ b, float* hg) {
  const int t = blockIdx.x, c = threadIdx.x;
  hg[t * 256 + c] = fmaxf(agg[t * 256 + c] + b[c], 0.0f);
}
__global__ void k_counts(const int* __restrict__ batch, float* counts) {
  const int b = threadIdx.x;  // 8 threads
  float cnt = 0.0f;
  for (int t = 0; t < NND; ++t) cnt += (batch[t] == b) ? 1.0f : 0.0f;
  counts[b] = fmaxf(cnt, 1.0f);
}
__global__ void k_hcat(const float* __restrict__ hg2, const float* __restrict__ att, float* hcat) {
  const int idx = blockIdx.x * 256 + threadIdx.x;
  const int t = idx >> 9, c = idx & 511;
  hcat[idx] = (c < 256) ? hg2[t * 256 + c] : att[t * 256 + c - 256];
}
__global__ void k_gate2(const float* __restrict__ gt1, const float* __restrict__ w2,
                        const float* __restrict__ b2, float* gvec) {
  __shared__ float red[256];
  const int t = blockIdx.x, c = threadIdx.x;
  red[c] = gt1[t * 256 + c] * w2[c];
  __syncthreads();
  for (int s = 128; s > 0; s >>= 1) { if (c < s) red[c] += red[c + s]; __syncthreads(); }
  if (c == 0) gvec[t] = sigmf(red[0] + b2[0]);
}
__global__ void k_zero(float* p, int n) {
  const int i = blockIdx.x * blockDim.x + threadIdx.x;
  if (i < n) p[i] = 0.0f;
}
__global__ void k_pool_acc(const int* __restrict__ batch, const float* __restrict__ hg2,
                           const float* __restrict__ att, const float* __restrict__ gvec, float* pooled) {
  const int t = blockIdx.x, c = threadIdx.x;
  atomicAdd(&pooled[batch[t] * 256 + c], hg2[t * 256 + c] + gvec[t] * att[t * 256 + c]);
}
__global__ void k_pool_fin(float* pooled, const float* __restrict__ counts, float* sbuf, int i) {
  const int b = blockIdx.x, c = threadIdx.x;
  const float v = pooled[b * 256 + c] / counts[b];
  pooled[b * 256 + c] = v;
  sbuf[(b * NC + i) * 256 + c] = v;
}
__global__ void k_pred(const float* __restrict__ pooled, const float* __restrict__ w1,
                       const float* __restrict__ b1, const float* __restrict__ w2,
                       const float* __restrict__ b2, float* sc) {
  __shared__ float sh[64];
  const int b = blockIdx.x, u = threadIdx.x;   // 64 threads
  float a = b1[u];
  for (int k = 0; k < 256; ++k) a += pooled[b * 256 + k] * w1[u * 256 + k];
  a = fmaxf(a, 0.0f);
  sh[u] = a * w2[u];
  __syncthreads();
  if (u == 0) {
    float s = b2[0];
    for (int k = 0; k < 64; ++k) s += sh[k];
    sc[b] = sigmf(s);
  }
}

// ---------------- tiny transformer over s[8][6][256] -----------------------
__global__ void k_tr_qkv(const float* __restrict__ sbuf, const float* __restrict__ w,
                         const float* __restrict__ b, float* qkv6) {
  const int r = blockIdx.x;                         // j*8+t
  const int col = blockIdx.y * 256 + threadIdx.x;   // 0..767
  const int j = r >> 3, t = r & 7;
  const float* row = sbuf + (t * NC + j) * 256;
  float acc = b[col];
  for (int k = 0; k < 256; ++k) acc += row[k] * w[col * 256 + k];
  qkv6[r * 768 + col] = acc;
}
__global__ void k_tr_attn(const float* __restrict__ qkv6, float* tatt) {
  const int j = blockIdx.x, hh = blockIdx.y, t = threadIdx.x;  // 8 threads
  const float* q = qkv6 + (j * 8 + t) * 768 + hh * 64;
  float sc[8];
  float mx = -1e30f;
  for (int u = 0; u < 8; ++u) {
    const float* kk = qkv6 + (j * 8 + u) * 768 + 256 + hh * 64;
    float s = 0.0f;
    for (int d = 0; d < 64; ++d) s += q[d] * kk[d];
    s *= 0.125f;
    sc[u] = s;
    mx = fmaxf(mx, s);
  }
  float se = 0.0f;
  for (int u = 0; u < 8; ++u) { sc[u] = expf(sc[u] - mx); se += sc[u]; }
  const float inv = 1.0f / se;
  for (int d = 0; d < 64; ++d) {
    float o = 0.0f;
    for (int u = 0; u < 8; ++u) o += sc[u] * qkv6[(j * 8 + u) * 768 + 512 + hh * 64 + d];
    tatt[(j * 8 + t) * 256 + hh * 64 + d] = o * inv;
  }
}
__global__ void k_tr_proj(const float* __restrict__ tatt, const float* __restrict__ w,
                          const float* __restrict__ b, float* tsa) {
  const int r = blockIdx.x, c = threadIdx.x;
  float acc = b[c];
  for (int k = 0; k < 256; ++k) acc += tatt[r * 256 + k] * w[c * 256 + k];
  tsa[r * 256 + c] = acc;
}
__global__ void k_tr_addln(float* sbuf, const float* __restrict__ delta,
                           const float* __restrict__ g, const float* __restrict__ bb) {
  __shared__ float red[256];
  const int r = blockIdx.x;
  const int j = r >> 3, t = r & 7;
  const int c = threadIdx.x;
  const int srow = (t * NC + j) * 256;
  const float x = sbuf[srow + c] + delta[r * 256 + c];
  red[c] = x; __syncthreads();
  for (int s = 128; s > 0; s >>= 1) { if (c < s) red[c] += red[c + s]; __syncthreads(); }
  const float mean = red[0] * (1.0f / 256.0f);
  __syncthreads();
  const float dx = x - mean;
  red[c] = dx * dx; __syncthreads();
  for (int s = 128; s > 0; s >>= 1) { if (c < s) red[c] += red[c + s]; __syncthreads(); }
  const float var = red[0] * (1.0f / 256.0f);
  sbuf[srow + c] = dx * rsqrtf(var + 1e-5f) * g[c] + bb[c];
}
__global__ void k_tr_ff1(const float* __restrict__ sbuf, const float* __restrict__ w,
                         const float* __restrict__ b, float* tff) {
  const int r = blockIdx.x;
  const int col = blockIdx.y * 256 + threadIdx.x;  // 0..2047
  const int j = r >> 3, t = r & 7;
  const float* row = sbuf + (t * NC + j) * 256;
  float acc = b[col];
  for (int k = 0; k < 256; ++k) acc += row[k] * w[col * 256 + k];
  tff[r * 2048 + col] = fmaxf(acc, 0.0f);
}
__global__ void k_tr_ff2(const float* __restrict__ tff, const float* __restrict__ w,
                         const float* __restrict__ b, float* tsa) {
  const int r = blockIdx.x, c = threadIdx.x;
  float acc = b[c];
  for (int k = 0; k < 2048; ++k) acc += tff[r * 2048 + k] * w[c * 2048 + k];
  tsa[r * 256 + c] = acc;
}
__global__ void k_final(const float* __restrict__ sbuf, const float* __restrict__ sc, float* out) {
  __shared__ float red[256];
  const int b = blockIdx.x, i = blockIdx.y, c = threadIdx.x;
  red[c] = sbuf[(b * NC + i) * 256 + c];
  __syncthreads();
  for (int s = 128; s > 0; s >>= 1) { if (c < s) red[c] += red[c + s]; __syncthreads(); }
  if (c == 0) {
    const float f = sigmf(red[0] * (1.0f / 256.0f));
    out[i * NB + b] = sc[i * NB + b] * (1.0f + 0.2f * f);
  }
}

// ============================================================================
// Host launcher
// ============================================================================
extern "C" void kernel_launch(void* const* d_in, const int* in_sizes, int n_in,
                              void* d_out, int out_size, void* d_ws, size_t ws_size,
                              hipStream_t stream)
{
  (void)n_in; (void)out_size; (void)ws_size;
  const float* x      = (const float*)d_in[0];
  const int*   ei     = (const int*)d_in[1];
  const int*   batch  = (const int*)d_in[2];
  const float* wif = (const float*)d_in[3];
  const float* whf = (const float*)d_in[4];
  const float* bif = (const float*)d_in[5];
  const float* bhf = (const float*)d_in[6];
  const float* wir = (const float*)d_in[7];
  const float* whr = (const float*)d_in[8];
  const float* bir = (const float*)d_in[9];
  const float* bhr = (const float*)d_in[10];
  const float* gcn1_w = (const float*)d_in[11];
  const float* gcn1_b = (const float*)d_in[12];
  const float* gcn2_w = (const float*)d_in[13];
  const float* gcn2_b = (const float*)d_in[14];
  const float* attn_in_w  = (const float*)d_in[15];
  const float* attn_in_b  = (const float*)d_in[16];
  const float* attn_out_w = (const float*)d_in[17];
  const float* attn_out_b = (const float*)d_in[18];
  const float* gate_w1 = (const float*)d_in[19];
  const float* gate_b1 = (const float*)d_in[20];
  const float* gate_w2 = (const float*)d_in[21];
  const float* gate_b2 = (const float*)d_in[22];
  const float* pred_w1 = (const float*)d_in[23];
  const float* pred_b1 = (const float*)d_in[24];
  const float* pred_w2 = (const float*)d_in[25];
  const float* pred_b2 = (const float*)d_in[26];
  const float* tr_in_w  = (const float*)d_in[27];
  const float* tr_in_b  = (const float*)d_in[28];
  const float* tr_out_w = (const float*)d_in[29];
  const float* tr_out_b = (const float*)d_in[30];
  const float* tr_ff1_w = (const float*)d_in[31];
  const float* tr_ff1_b = (const float*)d_in[32];
  const float* tr_ff2_w = (const float*)d_in[33];
  const float* tr_ff2_b = (const float*)d_in[34];
  const float* tr_ln1_g = (const float*)d_in[35];
  const float* tr_ln1_b = (const float*)d_in[36];
  const float* tr_ln2_g = (const float*)d_in[37];
  const float* tr_ln2_b = (const float*)d_in[38];

  const int E = in_sizes[1] / 2;
  float* ws = (float*)d_ws;
  float* out = (float*)d_out;

  float* pre_f  = ws;                       // 3145728
  float* pre_r  = ws + 3145728;             // 3145728
  float* hlstm  = ws + 6291456;             // 1572864
  float* whhT   = ws + 7864320;             // 524288
  float* bsum   = ws + 8388608;             // 2048
  float* deg    = ws + 8390656;             // 3072 (holds dinv after rsqrt)
  float* xw     = ws + 8393728;             // 786432
  float* agg    = ws + 9180160;             // 786432
  float* hg1    = ws + 9966592;             // 786432
  float* hg2    = ws + 10753024;            // 786432
  float* qkv    = ws + 11539456;            // 2359296
  float* oat    = ws + 13898752;            // 786432
  float* att    = ws + 14685184;            // 786432
  float* hcat   = ws + 15471616;            // 1572864
  float* gt1    = ws + 17044480;            // 786432
  float* gvec   = ws + 17830912;            // 3072
  float* pooled = ws + 17833984;            // 2048
  float* counts = ws + 17836032;            // 8
  float* scores = ws + 17836040;            // 48
  float* sbuf   = ws + 17836088;            // 12288
  float* tqkv   = ws + 17848376;            // 36864
  float* tsa    = ws + 17885240;            // 12288
  float* tff    = ws + 17897528;            // 98304
  float* tatt   = ws + 17995832;            // 12288

  // ---- BiLSTM ----
  k_bias_sum<<<8, 256, 0, stream>>>(bif, bhf, bir, bhr, bsum);
  k_transpose_whh<<<2048, 256, 0, stream>>>(whf, whr, whhT);
  k_wmma_gemm<<<dim3(192, 16), 32, 0, stream>>>(x, wif, bsum,        nullptr, pre_f, NND, 1024, 768, 0);
  k_wmma_gemm<<<dim3(192, 16), 32, 0, stream>>>(x, wir, bsum + 1024, nullptr, pre_r, NND, 1024, 768, 0);
  k_lstm<<<2, 1024, 0, stream>>>(pre_f, pre_r, whhT, hlstm);

  // ---- degree / normalization (shared by both GCN layers) ----
  k_deg_init<<<12, 256, 0, stream>>>(deg);
  k_deg_count<<<(E + 255) / 256, 256, 0, stream>>>(ei, E, deg);
  k_deg_rsqrt<<<12, 256, 0, stream>>>(deg);

  // ---- GCN layer 1 (512 -> 256) ----
  k_wmma_gemm<<<dim3(192, 4), 32, 0, stream>>>(hlstm, gcn1_w, nullptr, nullptr, xw, NND, 256, 512, 0);
  k_agg_self<<<NND, 256, 0, stream>>>(xw, deg, agg);
  k_agg_edges<<<E, 256, 0, stream>>>(ei, E, deg, xw, agg);
  k_gcn_fin<<<NND, 256, 0, stream>>>(agg, gcn1_b, hg1);

  // ---- GCN layer 2 (256 -> 256) ----
  k_wmma_gemm<<<dim3(192, 4), 32, 0, stream>>>(hg1, gcn2_w, nullptr, nullptr, xw, NND, 256, 256, 0);
  k_agg_self<<<NND, 256, 0, stream>>>(xw, deg, agg);
  k_agg_edges<<<E, 256, 0, stream>>>(ei, E, deg, xw, agg);
  k_gcn_fin<<<NND, 256, 0, stream>>>(agg, gcn2_b, hg2);

  k_counts<<<1, 8, 0, stream>>>(batch, counts);

  // ---- per-risk attention / gate / pool / predictor ----
  for (int i = 0; i < NC; ++i) {
    const float* aiw = attn_in_w  + (size_t)i * 768 * 256;
    const float* aib = attn_in_b  + (size_t)i * 768;
    const float* aow = attn_out_w + (size_t)i * 256 * 256;
    const float* aob = attn_out_b + (size_t)i * 256;
    const float* gw1 = gate_w1 + (size_t)i * 256 * 512;
    const float* gb1 = gate_b1 + (size_t)i * 256;
    const float* gw2 = gate_w2 + (size_t)i * 256;
    const float* gb2 = gate_b2 + i;
    const float* pw1 = pred_w1 + (size_t)i * 64 * 256;
    const float* pb1 = pred_b1 + (size_t)i * 64;
    const float* pw2 = pred_w2 + (size_t)i * 64;
    const float* pb2 = pred_b2 + i;

    k_wmma_gemm<<<dim3(192, 12), 32, 0, stream>>>(hg2, aiw, aib, nullptr, qkv, NND, 768, 256, 0);
    k_flash_attn<<<dim3(48, 4), 128, 0, stream>>>(qkv, oat);
    k_wmma_gemm<<<dim3(192, 4), 32, 0, stream>>>(oat, aow, aob, nullptr, att, NND, 256, 256, 0);
    k_hcat<<<NND * 2, 256, 0, stream>>>(hg2, att, hcat);
    k_wmma_gemm<<<dim3(192, 4), 32, 0, stream>>>(hcat, gw1, gb1, nullptr, gt1, NND, 256, 512, 1);
    k_gate2<<<NND, 256, 0, stream>>>(gt1, gw2, gb2, gvec);
    k_zero<<<8, 256, 0, stream>>>(pooled, 2048);
    k_pool_acc<<<NND, 256, 0, stream>>>(batch, hg2, att, gvec, pooled);
    k_pool_fin<<<NB, 256, 0, stream>>>(pooled, counts, sbuf, i);
    k_pred<<<NB, 64, 0, stream>>>(pooled, pw1, pb1, pw2, pb2, scores + i * NB);
  }

  // ---- 2-layer post-LN transformer over s[8][6][256] ----
  for (int l = 0; l < 2; ++l) {
    const float* iw  = tr_in_w  + (size_t)l * 768 * 256;
    const float* ib  = tr_in_b  + (size_t)l * 768;
    const float* ow  = tr_out_w + (size_t)l * 256 * 256;
    const float* ob  = tr_out_b + (size_t)l * 256;
    const float* f1w = tr_ff1_w + (size_t)l * 2048 * 256;
    const float* f1b = tr_ff1_b + (size_t)l * 2048;
    const float* f2w = tr_ff2_w + (size_t)l * 256 * 2048;
    const float* f2b = tr_ff2_b + (size_t)l * 256;

    k_tr_qkv<<<dim3(48, 3), 256, 0, stream>>>(sbuf, iw, ib, tqkv);
    k_tr_attn<<<dim3(NC, 4), 8, 0, stream>>>(tqkv, tatt);
    k_tr_proj<<<48, 256, 0, stream>>>(tatt, ow, ob, tsa);
    k_tr_addln<<<48, 256, 0, stream>>>(sbuf, tsa, tr_ln1_g + l * 256, tr_ln1_b + l * 256);
    k_tr_ff1<<<dim3(48, 8), 256, 0, stream>>>(sbuf, f1w, f1b, tff);
    k_tr_ff2<<<48, 256, 0, stream>>>(tff, f2w, f2b, tsa);
    k_tr_addln<<<48, 256, 0, stream>>>(sbuf, tsa, tr_ln2_g + l * 256, tr_ln2_b + l * 256);
  }

  k_final<<<dim3(NB, NC), 256, 0, stream>>>(sbuf, scores, out);
}
